// MultilayerModulatedDeformableC3D_53300544143737
// MI455X (gfx1250) — matrix operand order repaired and checked by reading, not compiled, not run on hardware
//
#include <hip/hip_runtime.h>
#include <hip/hip_bf16.h>
#include <math.h>

// ---------------------------------------------------------------------------
// MultilayerModulatedDeformableC3D for MI455X (gfx1250, wave32, WMMA).
//
// Decomposition per stage (B=1, C=256, D=8, H=32, W=32, N=8192, K=C*27=6912):
//   1) cvtA:      fp32 weights -> bf16 A matrix [Mpad=256][6912] (rows >= M zeroed)
//   2) im2col:    input fp32 -> bf16 B matrix [N=8192][K=6912] (K-contiguous)
//   3) gemm_wmma: d[108][8192] = A x B + bias           (v_wmma_f32_16x16x32_bf16)
//   4) sample:    trilinear-deform gather * sigmoid(mask) -> bf16 B [8192][6912]
//   5) cvtA + gemm_wmma: out[256][8192] = W x B + bias (+ReLU)
// Repeated 3x, then the final stage uses offsets from h3 but samples original x.
//
// GEMM inner loop is register double-buffered with STATIC indices (k-loop
// unrolled by 2: a0/b0[] for even 32-slices, a1/b1[] for odd ones) so the
// ping-pong stays in VGPRs — no dynamically-indexed aggregates, no scratch.
// ---------------------------------------------------------------------------

typedef __bf16 v16bf __attribute__((ext_vector_type(16)));
typedef float  v8f   __attribute__((ext_vector_type(8)));

#define NVOX  8192          // D*H*W = 8*32*32
#define KTOT  6912          // 256 * 27
#define MPAD  256           // A buffer always padded to 256 rows

__device__ __forceinline__ unsigned short f2bf(float f) {
    union { float f; unsigned int u; } cv; cv.f = f;
    unsigned int u = cv.u;
    // round-to-nearest-even fp32 -> bf16
    return (unsigned short)((u + 0x7FFFu + ((u >> 16) & 1u)) >> 16);
}

// ---- weight fp32 [M][KTOT] -> bf16 A [MPAD][KTOT], rows >= M zero-filled ----
__global__ void cvtA_kernel(const float* __restrict__ w,
                            unsigned short* __restrict__ Abuf, int M) {
    int t = blockIdx.x * blockDim.x + threadIdx.x;      // MPAD*KTOT threads
    if (t >= MPAD * KTOT) return;
    int row = t / KTOT;
    Abuf[t] = (row < M) ? f2bf(w[t]) : (unsigned short)0;
}

// ---- im2col: in fp32 [256][8][32][32] -> B bf16 [NVOX][KTOT] (zero pad) ----
__global__ void im2col_kernel(const float* __restrict__ in,
                              unsigned short* __restrict__ Bm) {
    int t = blockIdx.x * blockDim.x + threadIdx.x;      // 256 * NVOX threads
    if (t >= 256 * NVOX) return;
    int n = t & (NVOX - 1);                             // voxel (coalesced reads)
    int c = t >> 13;                                    // channel
    int wq = n & 31, hq = (n >> 5) & 31, dq = n >> 10;
    const float* inc = in + (size_t)c * NVOX;
    unsigned short* dst = Bm + (size_t)n * KTOT + c * 27;
    #pragma unroll
    for (int kd = 0; kd < 3; ++kd)
      #pragma unroll
      for (int kh = 0; kh < 3; ++kh)
        #pragma unroll
        for (int kw = 0; kw < 3; ++kw) {
            int z = dq + kd - 1, y = hq + kh - 1, x = wq + kw - 1;
            float v = 0.f;
            if ((unsigned)z < 8u && (unsigned)y < 32u && (unsigned)x < 32u)
                v = inc[(z * 32 + y) * 32 + x];
            *dst++ = f2bf(v);
        }
}

// ---- modulated deformable trilinear sampling -> B bf16 [NVOX][KTOT] --------
// d layout: [108][NVOX]; channels k*3+{0,1,2} = offsets (d,h,w), 81+k = mask.
__global__ void deform_sample_kernel(const float* __restrict__ in,
                                     const float* __restrict__ d,
                                     unsigned short* __restrict__ Bm) {
    int t = blockIdx.x * blockDim.x + threadIdx.x;      // 27 * NVOX threads
    if (t >= 27 * NVOX) return;
    int n = t & (NVOX - 1);
    int k = t >> 13;
    int wq = n & 31, hq = (n >> 5) & 31, dq = n >> 10;
    int kd = k / 9 - 1, kh = (k / 3) % 3 - 1, kw = k % 3 - 1;

    float cd = (float)(dq + kd) + d[(size_t)(k * 3 + 0) * NVOX + n];
    float ch = (float)(hq + kh) + d[(size_t)(k * 3 + 1) * NVOX + n];
    float cw = (float)(wq + kw) + d[(size_t)(k * 3 + 2) * NVOX + n];
    float msk = 1.f / (1.f + __expf(-d[(size_t)(81 + k) * NVOX + n]));

    float fd0 = floorf(cd), fh0 = floorf(ch), fw0 = floorf(cw);
    float fd = cd - fd0, fh = ch - fh0, fw = cw - fw0;
    int z0 = (int)fd0, y0 = (int)fh0, x0 = (int)fw0;

    float wgt[8];
    int   idx[8];
    #pragma unroll
    for (int i = 0; i < 8; ++i) {
        int dz = i >> 2, dy = (i >> 1) & 1, dx = i & 1;
        int z = z0 + dz, y = y0 + dy, x = x0 + dx;
        float wv = (dz ? fd : 1.f - fd) * (dy ? fh : 1.f - fh) * (dx ? fw : 1.f - fw);
        bool ok = (unsigned)z < 8u && (unsigned)y < 32u && (unsigned)x < 32u;
        wgt[i] = ok ? wv * msk : 0.f;
        int zc = z < 0 ? 0 : (z > 7 ? 7 : z);
        int yc = y < 0 ? 0 : (y > 31 ? 31 : y);
        int xc = x < 0 ? 0 : (x > 31 ? 31 : x);
        idx[i] = (zc * 32 + yc) * 32 + xc;
    }

    unsigned short* dst = Bm + (size_t)n * KTOT + k;
    for (int c = 0; c < 256; ++c) {
        const float* inc = in + (size_t)c * NVOX;
        float acc = 0.f;
        #pragma unroll
        for (int i = 0; i < 8; ++i) acc = fmaf(wgt[i], inc[idx[i]], acc);
        dst[c * 27] = f2bf(acc);
    }
}

// ---- bf16 WMMA GEMM: out[M][NVOX] = A[MPAD][KTOT] x B[NVOX][KTOT]^T --------
// Block = 8 waves: 2 m-tiles x 4 n-groups; wave computes 16(M) x 128(N).
struct Frag { uint4 u0, u1; };                          // 32B = 16 bf16

__device__ __forceinline__ v16bf frag_v(const Frag& f) {
    union { Frag f; v16bf v; } c; c.f = f; return c.v;
}

__device__ __forceinline__ void load_a(Frag& f, const unsigned short* Arow, int k) {
    f.u0 = *reinterpret_cast<const uint4*>(Arow + k);
    f.u1 = *reinterpret_cast<const uint4*>(Arow + k + 16);
}
__device__ __forceinline__ void load_b(Frag& f, const unsigned short* Bcol0,
                                       int j, int k) {
    const unsigned short* Bp = Bcol0 + (size_t)(j * 16) * KTOT + k;
    f.u0 = *reinterpret_cast<const uint4*>(Bp);
    f.u1 = *reinterpret_cast<const uint4*>(Bp + 16);
}

__global__ __launch_bounds__(256, 1) void gemm_wmma_kernel(
        const unsigned short* __restrict__ A,
        const unsigned short* __restrict__ B,
        const float* __restrict__ bias,
        float* __restrict__ out, int M, int relu) {
    int lane   = threadIdx.x & 31;
    int wave   = threadIdx.x >> 5;
    int laneLo = lane & 15;
    int laneHi = lane >> 4;                                // 0 or 1
    int mTile  = (blockIdx.y * 2 + (wave & 1)) * 16;       // 16-row tile base
    int nBase  = blockIdx.x * 512 + (wave >> 1) * 128;     // 8 x 16-col tiles

    v8f acc[8];
    #pragma unroll
    for (int j = 0; j < 8; ++j) acc[j] = (v8f){0.f,0.f,0.f,0.f,0.f,0.f,0.f,0.f};

    // A fragment (16x32 bf16): lane holds row m = mTile+laneLo,
    // K halves {b..b+7} U {16+b..16+b+7}, b = 8*laneHi  -> two 16B loads.
    const unsigned short* Arow = A + (size_t)(mTile + laneLo) * KTOT + laneHi * 8;
    // B fragment (32x16 bf16): lane holds col n, same K halves as A -> two 16B loads.
    const unsigned short* Bcol0 = B + (size_t)(nBase + laneLo) * KTOT + laneHi * 8;

    Frag a0, a1, b0[8], b1[8];                             // static indices only

    load_a(a0, Arow, 0);
    #pragma unroll
    for (int j = 0; j < 8; ++j) load_b(b0[j], Bcol0, j, 0);

    int k = 0;
    for (; k < KTOT - 64; k += 64) {
        // fill odd buffer (slice k+32) before computing on even buffer (slice k)
        load_a(a1, Arow, k + 32);
        #pragma unroll
        for (int j = 0; j < 8; ++j) load_b(b1[j], Bcol0, j, k + 32);
        __builtin_prefetch(Arow + k + 256, 0, 1);          // global_prefetch_b8
        #pragma unroll
        for (int j = 0; j < 8; ++j)
            acc[j] = __builtin_amdgcn_wmma_f32_16x16x32_bf16(
                false, frag_v(a0), false, frag_v(b0[j]), (short)0, acc[j],
                false, false);
        // fill even buffer (slice k+64) before computing on odd buffer (k+32)
        load_a(a0, Arow, k + 64);
        #pragma unroll
        for (int j = 0; j < 8; ++j) load_b(b0[j], Bcol0, j, k + 64);
        #pragma unroll
        for (int j = 0; j < 8; ++j)
            acc[j] = __builtin_amdgcn_wmma_f32_16x16x32_bf16(
                false, frag_v(a1), false, frag_v(b1[j]), (short)0, acc[j],
                false, false);
    }
    // tail: k == KTOT-64; b0 holds slice k, load+compute final slice k+32
    load_a(a1, Arow, k + 32);
    #pragma unroll
    for (int j = 0; j < 8; ++j) load_b(b1[j], Bcol0, j, k + 32);
    #pragma unroll
    for (int j = 0; j < 8; ++j)
        acc[j] = __builtin_amdgcn_wmma_f32_16x16x32_bf16(
            false, frag_v(a0), false, frag_v(b0[j]), (short)0, acc[j],
            false, false);
    #pragma unroll
    for (int j = 0; j < 8; ++j)
        acc[j] = __builtin_amdgcn_wmma_f32_16x16x32_bf16(
            false, frag_v(a1), false, frag_v(b1[j]), (short)0, acc[j],
            false, false);

    // C/D layout: lane(0..15): n = nBase+16j+lane, rows m = r; lane(16..31): m = 8+r.
    #pragma unroll
    for (int j = 0; j < 8; ++j) {
        int n = nBase + j * 16 + laneLo;
        #pragma unroll
        for (int r = 0; r < 8; ++r) {
            int m = mTile + laneHi * 8 + r;
            if (m < M) {
                float v = acc[j][r] + bias[m];
                if (relu) v = fmaxf(v, 0.f);
                out[(size_t)m * NVOX + n] = v;
            }
        }
    }
}

// ---------------------------------------------------------------------------
extern "C" void kernel_launch(void* const* d_in, const int* in_sizes, int n_in,
                              void* d_out, int out_size, void* d_ws, size_t ws_size,
                              hipStream_t stream) {
    (void)in_sizes; (void)n_in; (void)out_size; (void)ws_size;
    const float* x = (const float*)d_in[0];
    const float* ow[3] = {(const float*)d_in[1], (const float*)d_in[5], (const float*)d_in[9]};
    const float* ob[3] = {(const float*)d_in[2], (const float*)d_in[6], (const float*)d_in[10]};
    const float* wt[3] = {(const float*)d_in[3], (const float*)d_in[7], (const float*)d_in[11]};
    const float* bi[3] = {(const float*)d_in[4], (const float*)d_in[8], (const float*)d_in[12]};
    const float* defo_w = (const float*)d_in[13];
    const float* defo_b = (const float*)d_in[14];
    const float* c3d_w  = (const float*)d_in[15];
    const float* c3d_b  = (const float*)d_in[16];
    float* outp = (float*)d_out;

    // Workspace carve-up (~131 MiB; GEMM B matrix sized for the 192MB L2).
    char* ws = (char*)d_ws;
    size_t off = 0;
    auto carve = [&](size_t bytes) {
        void* p = ws + off;
        off += (bytes + 255) & ~(size_t)255;
        return p;
    };
    unsigned short* Abuf = (unsigned short*)carve((size_t)MPAD * KTOT * 2);   // 3.5 MB
    unsigned short* Bbuf = (unsigned short*)carve((size_t)NVOX * KTOT * 2);   // 113 MB
    float* dbuf = (float*)carve((size_t)108 * NVOX * 4);                      // 3.5 MB
    float* h0   = (float*)carve((size_t)256 * NVOX * 4);                      // 8 MB
    float* h1   = (float*)carve((size_t)256 * NVOX * 4);                      // 8 MB

    const int T = 256;
    dim3 gCvt((MPAD * KTOT + T - 1) / T), gIm((256 * NVOX + T - 1) / T),
         gSm((27 * NVOX + T - 1) / T);
    dim3 gGemm108(NVOX / 512, 4);   // rows 0..127 computed, stores masked to M=108
    dim3 gGemm256(NVOX / 512, 8);

    // conv_in: source of the offsets conv; samp_in: tensor being deform-sampled.
    auto stage = [&](const float* conv_in, const float* samp_in,
                     const float* owp, const float* obp,
                     const float* wp, const float* bp,
                     float* outb, int relu) {
        cvtA_kernel<<<gCvt, T, 0, stream>>>(owp, Abuf, 108);
        im2col_kernel<<<gIm, T, 0, stream>>>(conv_in, Bbuf);
        gemm_wmma_kernel<<<gGemm108, T, 0, stream>>>(Abuf, Bbuf, obp, dbuf, 108, 0);
        deform_sample_kernel<<<gSm, T, 0, stream>>>(samp_in, dbuf, Bbuf);
        cvtA_kernel<<<gCvt, T, 0, stream>>>(wp, Abuf, 256);
        gemm_wmma_kernel<<<gGemm256, T, 0, stream>>>(Abuf, Bbuf, bp, outb, 256, relu);
    };

    stage(x,  x,  ow[0], ob[0], wt[0], bi[0], h0, 1);   // h1 = relu(mdconv(x))
    stage(h0, h0, ow[1], ob[1], wt[1], bi[1], h1, 1);   // h2 = relu(mdconv(h1))
    stage(h1, h1, ow[2], ob[2], wt[2], bi[2], h0, 1);   // h3 = relu(mdconv(h2))
    // Final: offsets/alpha from conv3d(h3, defo_w); deform conv samples ORIGINAL x.
    stage(h0, x, defo_w, defo_b, c3d_w, c3d_b, outp, 0);
}
